// MACRO_VRNN_75127567942022
// MI455X (gfx1250) — matrix operand order repaired and checked by reading, not compiled, not run on hardware
//
#include <hip/hip_runtime.h>

// ---------------------------------------------------------------------------
// MACRO_VRNN on MI455X (gfx1250): persistent cooperative kernel, round 3.
// - Wave-scalar tile scheduling via readfirstlane -> SALU tile decode, scalar
//   branches, SGPR-base addressing; no EXEC-masked WMMA regions.
// - All GEMMs have Np multiple of 32: zm/zs fused into one [Wpm|Wps] GEMM
//   (zc, Np=32), xm/xs fused into [Wdx|Wds] (xc, Np=32) -> branch-free 2x2
//   register-blocked inner loop (4 WMMAs per 2 A-frag + 2 B-frag loads).
// - All GEMM A-inputs bf16 (2x b128 loads/lane, zero conversion VALU);
//   v_wmma_f32_16x16x32_bf16 everywhere, f32 accumulate.
// - 63 sequential steps, ~15 dependent phases each, software device-wide
//   barrier across 64 resident workgroups.  Requires ws_size >= ~43MB.
// ---------------------------------------------------------------------------

#define T_ 64
#define B_ 512
#define A_ 5
#define XD_ 2
#define YD_ 10
#define ZD_ 16
#define HD_ 200
#define MD_ 90
#define HMIC_ 200
#define HMAC_ 200

// padded dims: K multiples of 32, N multiples of 32
#define HDP 224     // HD=200 padded
#define MDP 96      // MD=90 padded
#define ZDP 32
#define KDM 224     // YD+HMAC = 210
#define KPR 320     // MD+HMIC = 290
#define KD_IN 320   // YD+MD+ZD+HMIC = 316
#define KMIC 32     // XD+ZD = 18
#define KMAC 480    // A*MD = 450
#define G3 608      // 3*200 = 600 padded

#define NWG 64
#define BLK 256
#define PREP_BLOCKS 256

typedef __attribute__((ext_vector_type(16))) __bf16 v16bf;
typedef __attribute__((ext_vector_type(8)))  float  v8f;

// ----- float workspace region (element offsets) -----------------------------
constexpr size_t OFF_LOGITS = 0;
constexpr size_t OFF_M      = OFF_LOGITS + (size_t)A_*B_*MDP;
constexpr size_t OFF_ZC     = OFF_M      + (size_t)A_*B_*MDP;  // [A][B][32]: 0..15 mean, 16..31 std-logit
constexpr size_t OFF_Z      = OFF_ZC     + (size_t)A_*B_*32;
constexpr size_t OFF_XC     = OFF_Z      + (size_t)A_*B_*ZDP;  // [A][B][32]
constexpr size_t OFF_GIMAC  = OFF_XC     + (size_t)A_*B_*32;
constexpr size_t OFF_GHMAC  = OFF_GIMAC  + (size_t)B_*G3;
constexpr size_t OFF_GIMIC  = OFF_GHMAC  + (size_t)B_*G3;
constexpr size_t OFF_GHMIC  = OFF_GIMIC  + (size_t)A_*B_*G3;
constexpr size_t OFF_HMAC   = OFF_GHMIC  + (size_t)A_*B_*G3;   // [2][B][HDP]
constexpr size_t OFF_HMIC   = OFF_HMAC   + (size_t)2*B_*HDP;   // [2][A][B][HDP]
constexpr size_t OFF_YB     = OFF_HMIC   + (size_t)2*A_*B_*HDP;// [2][B][32]
constexpr size_t OFF_BIASZ  = OFF_YB     + (size_t)2*B_*32;    // [A][32] = [bpm|bps]
constexpr size_t OFF_BIASX  = OFF_BIASZ  + (size_t)A_*32;      // [A][32] = [bdx,0..|bds,0..]
constexpr size_t F_TOTAL    = OFF_BIASX  + (size_t)A_*32;

// ----- bf16 region: transposed weights [N x Kp] + bf16 activations ----------
constexpr size_t BO_WDM1   = 0;
constexpr size_t BO_WDM2   = BO_WDM1   + (size_t)A_*HDP*KDM;
constexpr size_t BO_WPR1   = BO_WDM2   + (size_t)A_*MDP*HDP;
constexpr size_t BO_WPR2   = BO_WPR1   + (size_t)A_*HDP*KPR;
constexpr size_t BO_WZC    = BO_WPR2   + (size_t)A_*HDP*HDP;   // [A][32][HDP] = [Wpm|Wps]
constexpr size_t BO_WD1    = BO_WZC    + (size_t)A_*32*HDP;
constexpr size_t BO_WD2    = BO_WD1    + (size_t)A_*HDP*KD_IN;
constexpr size_t BO_WXC    = BO_WD2    + (size_t)A_*HDP*HDP;   // [A][32][HDP] = [Wdx|Wds]
constexpr size_t BO_WIHMIC = BO_WXC    + (size_t)A_*32*HDP;
constexpr size_t BO_WHHMIC = BO_WIHMIC + (size_t)A_*G3*KMIC;
constexpr size_t BO_WIHMAC = BO_WHHMIC + (size_t)A_*G3*HDP;
constexpr size_t BO_WHHMAC = BO_WIHMAC + (size_t)G3*KMAC;
// bf16 activation buffers (GEMM A-inputs)
constexpr size_t AB_ACTDM  = BO_WHHMAC + (size_t)G3*HDP;
constexpr size_t AB_HDM    = AB_ACTDM  + (size_t)A_*B_*KDM;
constexpr size_t AB_ACTPR  = AB_HDM    + (size_t)A_*B_*HDP;
constexpr size_t AB_HP1    = AB_ACTPR  + (size_t)A_*B_*KPR;
constexpr size_t AB_HP     = AB_HP1    + (size_t)A_*B_*HDP;
constexpr size_t AB_ACTD   = AB_HP     + (size_t)A_*B_*HDP;
constexpr size_t AB_HD1    = AB_ACTD   + (size_t)A_*B_*KD_IN;
constexpr size_t AB_HDB    = AB_HD1    + (size_t)A_*B_*HDP;
constexpr size_t AB_ACTMAC = AB_HDB    + (size_t)A_*B_*HDP;
constexpr size_t AB_ACTMIC = AB_ACTMAC + (size_t)B_*KMAC;
constexpr size_t AB_HMACS  = AB_ACTMIC + (size_t)A_*B_*KMIC;  // [2][B][HDP]
constexpr size_t AB_HMICS  = AB_HMACS  + (size_t)2*B_*HDP;    // [2][A][B][HDP]
constexpr size_t B_TOTAL   = AB_HMICS  + (size_t)2*A_*B_*HDP;

struct Params {
  const float *y; const int *mg;
  const float *Wdm1,*bdm1,*Wdm2,*bdm2;
  const float *Wpr1,*bpr1,*Wpr2,*bpr2,*Wpm,*bpm,*Wps,*bps;
  const float *Wd1,*bd1,*Wd2,*bd2,*Wdx,*bdx,*Wds,*bds;
  const float *Wih_mic,*Whh_mic,*bih_mic,*bhh_mic;
  const float *Wih_mac,*Whh_mac,*bih_mac,*bhh_mac;
  const float *eps_g,*eps_z,*eps_x;
  const int *burn;
  float *out;
  float *wsf;
  unsigned short *wsb;   // bf16 storage (cast device-side)
  unsigned *bar;
};

// ----- small math helpers ---------------------------------------------------
__device__ __forceinline__ float sigmoidf_(float x) { return 1.f / (1.f + __expf(-x)); }
__device__ __forceinline__ float softplusf_(float x) { return (x > 20.f) ? x : log1pf(__expf(x)); }

// ----- device-wide barrier (all NWG workgroups resident) --------------------
__device__ __forceinline__ void gsync(unsigned* bar, int nwg) {
  __threadfence();
  __syncthreads();
  if (threadIdx.x == 0) {
    unsigned* arrive = bar;
    unsigned* gen    = bar + 1;
    unsigned g = __atomic_load_n(gen, __ATOMIC_RELAXED);
    unsigned prev = __atomic_fetch_add(arrive, 1u, __ATOMIC_ACQ_REL);
    if (prev == (unsigned)nwg - 1u) {
      __atomic_store_n(arrive, 0u, __ATOMIC_RELAXED);
      __atomic_fetch_add(gen, 1u, __ATOMIC_ACQ_REL);
    } else {
      while (__atomic_load_n(gen, __ATOMIC_ACQUIRE) == g)
        __builtin_amdgcn_s_sleep(2);
    }
  }
  __syncthreads();
}

// ----- WMMA fragment loads (CDNA5 16-bit layouts, wave32) -------------------
// A 16x32 bf16 (row-major [M][Kp]): lane L holds row M=L%16; halves 0..15 ->
// K = k0+sel16+0..15, sel16=(L>=16)?16:0.
__device__ __forceinline__ v16bf load_afrag(const __bf16* __restrict__ A, int ldk,
                                            int m0, int k0, int lane) {
  const int mr   = m0 + (lane & 15);
  const int sel8 = (lane & 16) ? 16 : 0;
  const __bf16* p = A + (size_t)mr * ldk + k0 + sel8;
  union { uint4 u[2]; v16bf v; } w;
  w.u[0] = *(const uint4*)(p);
  w.u[1] = *(const uint4*)(p + 8);
  return w.v;
}

// B 32x16 bf16 (stored transposed as Bt[N][Kp]): lane L holds col N=L%16;
// halves 0..15 -> K = k0 + (L/16)*16 + h  (16 contiguous bf16 = 2 x b128).
__device__ __forceinline__ v16bf load_bfrag(const __bf16* __restrict__ Bt, int ldk,
                                            int n0, int k0, int lane) {
  const int n    = n0 + (lane & 15);
  const int koff = k0 + ((lane >> 4) << 4);
  const uint4* pp = (const uint4*)(Bt + (size_t)n * ldk + koff);
  union { uint4 u[2]; v16bf v; } w;
  w.u[0] = pp[0];
  w.u[1] = pp[1];
  return w.v;
}

// C = A(512 x Kp, bf16) * Bt^T(Kp x Np, bf16) [+bias] [relu]; 2x2 tiles/wave.
// Np must be a multiple of 32 -> branch-free inner loop, EXEC all-ones.
// C/D f32 layout: lane L holds col N=L%16; vgpr v holds row M = m0+8*(L/16)+v.
template<bool OBF>
__device__ __forceinline__ void gemm_tiles(const __bf16* __restrict__ Aact, size_t aStr,
                                           const __bf16* __restrict__ Bt, size_t bStr,
                                           const float* __restrict__ bias, int biasStr,
                                           void* __restrict__ Cout, size_t cStr,
                                           int Np, int Kp, int realN, int nAg, bool relu,
                                           int wave, int nW, int lane) {
  const int ntg = Np >> 5;                // groups of 2 n-tiles (exact)
  const int mtg = B_ >> 5;                // 16 groups of 2 m-tiles
  const int groupsPer = mtg * ntg;
  const int total = nAg * groupsPer;
  for (int tl = wave; tl < total; tl += nW) {
    const int ag  = tl / groupsPer;
    const int r   = tl - ag * groupsPer;
    const int mgi = r / ntg;
    const int ngi = r - mgi * ntg;
    const int m0  = mgi << 5;
    const int n0  = ngi << 5;
    const __bf16* Ag = Aact + (size_t)ag * aStr;
    const __bf16* Bg = Bt   + (size_t)ag * bStr;
    float*  Cf = (float*)Cout  + (size_t)ag * cStr;
    __bf16* Cb = (__bf16*)Cout + (size_t)ag * cStr;

    v8f acc00 = {}, acc01 = {}, acc10 = {}, acc11 = {};
    for (int k0 = 0; k0 < Kp; k0 += 32) {
      v16bf a0 = load_afrag(Ag, Kp, m0,      k0, lane);
      v16bf a1 = load_afrag(Ag, Kp, m0 + 16, k0, lane);
      v16bf b0 = load_bfrag(Bg, Kp, n0,      k0, lane);
      v16bf b1 = load_bfrag(Bg, Kp, n0 + 16, k0, lane);
      acc00 = __builtin_amdgcn_wmma_f32_16x16x32_bf16(false, a0, false, b0,
                                                      (short)0, acc00, false, false);
      acc10 = __builtin_amdgcn_wmma_f32_16x16x32_bf16(false, a1, false, b0,
                                                      (short)0, acc10, false, false);
      acc01 = __builtin_amdgcn_wmma_f32_16x16x32_bf16(false, a0, false, b1,
                                                      (short)0, acc01, false, false);
      acc11 = __builtin_amdgcn_wmma_f32_16x16x32_bf16(false, a1, false, b1,
                                                      (short)0, acc11, false, false);
    }

    auto store_acc = [&](const v8f& acc, int mt0, int nt0) {
      const int n  = nt0 + (lane & 15);
      const int mb = mt0 + ((lane >> 4) << 3);
      float bb = 0.f;
      if (bias != nullptr && n < realN) bb = bias[(size_t)ag * biasStr + n];
#pragma unroll
      for (int v = 0; v < 8; ++v) {
        float val = acc[v] + bb;
        if (relu && val < 0.f) val = 0.f;
        if (OBF) Cb[(size_t)(mb + v) * Np + n] = (__bf16)val;
        else     Cf[(size_t)(mb + v) * Np + n] = val;
      }
    };
    store_acc(acc00, m0,      n0);
    store_acc(acc10, m0 + 16, n0);
    store_acc(acc01, m0,      n0 + 16);
    store_acc(acc11, m0 + 16, n0 + 16);
  }
}

// ----- weight transpose + pad to bf16 [nA][Np][Kp] --------------------------
// srcNK=false: src is [nA][K][N]; srcNK=true: src is [nA][N][K] (GRU weights)
__device__ __forceinline__ void trW(const float* __restrict__ src, __bf16* __restrict__ dst,
                                    int K, int N, int Kp, int Np, int nA, bool srcNK,
                                    int g, int gs) {
  const int per = Np * Kp;
  const int tot = nA * per;
  for (int i = g; i < tot; i += gs) {
    int a = i / per;
    int r = i - a * per;
    int n = r / Kp;
    int k = r - n * Kp;
    float v = 0.f;
    if (n < N && k < K)
      v = srcNK ? src[((size_t)a * N + n) * K + k]
                : src[((size_t)a * K + k) * N + n];
    dst[i] = (__bf16)v;
  }
}

// pair-fused: dst[a][n(32)][k(Kp)], n<16 from s0 [a][K][N], n>=16 from s1
__device__ __forceinline__ void trW_pair(const float* __restrict__ s0,
                                         const float* __restrict__ s1,
                                         __bf16* __restrict__ dst,
                                         int K, int N, int Kp, int nA,
                                         int g, int gs) {
  const int per = 32 * Kp;
  const int tot = nA * per;
  for (int i = g; i < tot; i += gs) {
    int a = i / per;
    int r = i - a * per;
    int n = r / Kp;
    int k = r - n * Kp;
    const float* s = (n < 16) ? s0 : s1;
    int nn = n & 15;
    float v = 0.f;
    if (nn < N && k < K) v = s[((size_t)a * K + k) * N + nn];
    dst[i] = (__bf16)v;
  }
}

// ----- prologue: weight conversion + state init -----------------------------
__global__ __launch_bounds__(BLK) void vrnn_prep(Params p) {
  const int g  = blockIdx.x * blockDim.x + threadIdx.x;
  const int gs = gridDim.x * blockDim.x;
  __bf16* wb = (__bf16*)p.wsb;
  float* wsf = p.wsf;

  trW(p.Wdm1,    wb + BO_WDM1,   YD_+HMAC_, HD_,  KDM,   HDP, A_, false, g, gs);
  trW(p.Wdm2,    wb + BO_WDM2,   HD_,       MD_,  HDP,   MDP, A_, false, g, gs);
  trW(p.Wpr1,    wb + BO_WPR1,   MD_+HMIC_, HD_,  KPR,   HDP, A_, false, g, gs);
  trW(p.Wpr2,    wb + BO_WPR2,   HD_,       HD_,  HDP,   HDP, A_, false, g, gs);
  trW(p.Wd1,     wb + BO_WD1,    316,       HD_,  KD_IN, HDP, A_, false, g, gs);
  trW(p.Wd2,     wb + BO_WD2,    HD_,       HD_,  HDP,   HDP, A_, false, g, gs);
  trW(p.Wih_mic, wb + BO_WIHMIC, XD_+ZD_,   600,  KMIC,  G3,  A_, true,  g, gs);
  trW(p.Whh_mic, wb + BO_WHHMIC, HMIC_,     600,  HDP,   G3,  A_, true,  g, gs);
  trW(p.Wih_mac, wb + BO_WIHMAC, A_*MD_,    600,  KMAC,  G3,  1,  true,  g, gs);
  trW(p.Whh_mac, wb + BO_WHHMAC, HMAC_,     600,  HDP,   G3,  1,  true,  g, gs);
  trW_pair(p.Wpm, p.Wps, wb + BO_WZC, HD_, ZD_, HDP, A_, g, gs);
  trW_pair(p.Wdx, p.Wds, wb + BO_WXC, HD_, XD_, HDP, A_, g, gs);

  // fused bias vectors (zeros in pad slots)
  for (int i = g; i < A_*32; i += gs) {
    int a = i / 32, n = i % 32, nn = n & 15;
    float vz = (n < 16) ? p.bpm[a*ZD_ + nn] : p.bps[a*ZD_ + nn];
    float vx = (nn < XD_) ? ((n < 16) ? p.bdx[a*XD_ + nn] : p.bds[a*XD_ + nn]) : 0.f;
    wsf[OFF_BIASZ + i] = vz;
    wsf[OFF_BIASX + i] = vx;
  }

  for (int i = g; i < 2*B_*HDP;    i += gs) wsf[OFF_HMAC + i] = 0.f;
  for (int i = g; i < 2*A_*B_*HDP; i += gs) wsf[OFF_HMIC + i] = 0.f;
  for (int i = g; i < 2*B_*HDP;    i += gs) wb[AB_HMACS + i] = (__bf16)0.f;
  for (int i = g; i < 2*A_*B_*HDP; i += gs) wb[AB_HMICS + i] = (__bf16)0.f;
  for (int i = g; i < 2*B_*32; i += gs) {
    int buf = i / (B_*32);
    int r   = i % (B_*32);
    int b = r / 32, k = r % 32;
    wsf[OFF_YB + i] = (buf == 0 && k < YD_) ? p.y[(size_t)b*YD_ + k] : 0.f;
  }
  for (int i = g; i < B_*YD_; i += gs) p.out[i] = p.y[i];   // ret[0] = y[0]
  if (g < 4) p.bar[g] = 0u;
}

// ----- main persistent kernel ----------------------------------------------
__global__ __launch_bounds__(BLK) void vrnn_main(Params p) {
  const int tid  = threadIdx.x;
  const int g    = blockIdx.x * BLK + tid;
  const int gs   = gridDim.x * BLK;
  // scalarize the wave id so tile scheduling is SALU + scalar branches
  const int wave = __builtin_amdgcn_readfirstlane(g >> 5);
  const int nW   = gs >> 5;
  const int lane = tid & 31;
  const int nwg  = gridDim.x;
  const int burn = p.burn[0];

  float* wsf    = p.wsf;
  float* logits = wsf + OFF_LOGITS;
  float* mbuf   = wsf + OFF_M;
  float* zc     = wsf + OFF_ZC;
  float* zbuf   = wsf + OFF_Z;
  float* xc     = wsf + OFF_XC;
  float* gimac  = wsf + OFF_GIMAC;
  float* ghmac  = wsf + OFF_GHMAC;
  float* gimic  = wsf + OFF_GIMIC;
  float* ghmic  = wsf + OFF_GHMIC;
  float* hmacB  = wsf + OFF_HMAC;
  float* hmicB  = wsf + OFF_HMIC;
  float* ybB    = wsf + OFF_YB;
  float* biasZ  = wsf + OFF_BIASZ;
  float* biasX  = wsf + OFF_BIASX;

  __bf16* wb     = (__bf16*)p.wsb;
  __bf16* actdm  = wb + AB_ACTDM;
  __bf16* hdm    = wb + AB_HDM;
  __bf16* actpr  = wb + AB_ACTPR;
  __bf16* hp1    = wb + AB_HP1;
  __bf16* hp     = wb + AB_HP;
  __bf16* actd   = wb + AB_ACTD;
  __bf16* hd1    = wb + AB_HD1;
  __bf16* hdb    = wb + AB_HDB;
  __bf16* actmac = wb + AB_ACTMAC;
  __bf16* actmic = wb + AB_ACTMIC;
  __bf16* hmacSB = wb + AB_HMACS;
  __bf16* hmicSB = wb + AB_HMICS;

  float* outGoals = p.out + (size_t)T_*B_*YD_;
  unsigned* bar = p.bar;

  for (int t = 0; t < T_ - 1; ++t) {
    const int cur = t & 1, nxt = cur ^ 1;
    float*  hmac   = hmacB + (size_t)cur * B_ * HDP;
    float*  hmacN  = hmacB + (size_t)nxt * B_ * HDP;
    float*  hmic   = hmicB + (size_t)cur * A_ * B_ * HDP;
    float*  hmicN  = hmicB + (size_t)nxt * A_ * B_ * HDP;
    __bf16* hmacS  = hmacSB + (size_t)cur * B_ * HDP;
    __bf16* hmacNS = hmacSB + (size_t)nxt * B_ * HDP;
    __bf16* hmicS  = hmicSB + (size_t)cur * A_ * B_ * HDP;
    __bf16* hmicNS = hmicSB + (size_t)nxt * A_ * B_ * HDP;
    float*  yb     = ybB + (size_t)cur * B_ * 32;
    float*  ybN    = ybB + (size_t)nxt * B_ * 32;

    // P1: actdm = [y | h_mac] (padded, bf16)
    for (int i = g; i < A_*B_*KDM; i += gs) {
      int r = i % (B_*KDM);
      int b = r / KDM, k = r % KDM;
      float v = 0.f;
      if (k < YD_)             v = yb[b*32 + k];
      else if (k < YD_+HMAC_)  v = hmac[(size_t)b*HDP + (k - YD_)];
      actdm[i] = (__bf16)v;
    }
    gsync(bar, nwg);

    // P2: hdm = relu(actdm @ Wdm1 + bdm1)  [bf16 out]
    gemm_tiles<true>(actdm, (size_t)B_*KDM, wb + BO_WDM1, (size_t)HDP*KDM, p.bdm1, HD_,
                     hdm, (size_t)B_*HDP, HDP, KDM, HD_, A_, true, wave, nW, lane);
    gsync(bar, nwg);

    // P3: logits = hdm @ Wdm2 + bdm2  (log_softmax elided: argmax-invariant)
    gemm_tiles<false>(hdm, (size_t)B_*HDP, wb + BO_WDM2, (size_t)MDP*HDP, p.bdm2, MD_,
                      logits, (size_t)B_*MDP, MDP, HDP, MD_, A_, false, wave, nW, lane);
    gsync(bar, nwg);

    // P4: gumbel-argmax sample / fixed one-hot; write m one-hot + goals output
    for (int i = g; i < A_*B_; i += gs) {
      int a = i / B_, b = i % B_;
      int curg = p.mg[((size_t)t*B_ + 0)*A_ + a];
      bool samp = (curg == -1) || ((t >= burn) && (burn > 0));
      float* mrow = mbuf + ((size_t)a*B_ + b)*MDP;
      int goal;
      if (samp) {
        const float* lg = logits + ((size_t)a*B_ + b)*MDP;
        const float* gg = p.eps_g + (((size_t)t*A_ + a)*B_ + b)*MD_;
        int best = 0; float bv = lg[0] + gg[0];
        for (int md = 1; md < MD_; ++md) {
          float v = lg[md] + gg[md];
          if (v > bv) { bv = v; best = md; }
        }
        goal = best;
        for (int md = 0; md < MDP; ++md) mrow[md] = (md == best) ? 1.f : 0.f;
      } else if (b == 0) {
        goal = curg;
        for (int md = 0; md < MDP; ++md) mrow[md] = (md == curg) ? 1.f : 0.f;
      } else {
        goal = 0;
        for (int md = 0; md < MDP; ++md) mrow[md] = 0.f;
      }
      outGoals[((size_t)t*B_ + b)*A_ + a] = (float)goal;
      if (t == T_ - 2) outGoals[((size_t)(T_-1)*B_ + b)*A_ + a] = (float)goal;
    }
    gsync(bar, nwg);

    // P5: actpr = [m | h_mic];  actmac = m transposed to (B, A*MD)  (bf16)
    for (int i = g; i < A_*B_*KPR; i += gs) {
      int a = i / (B_*KPR); int r = i % (B_*KPR);
      int b = r / KPR, k = r % KPR;
      float v = 0.f;
      if (k < MD_)            v = mbuf[((size_t)a*B_ + b)*MDP + k];
      else if (k < MD_+HMIC_) v = hmic[((size_t)a*B_ + b)*HDP + (k - MD_)];
      actpr[i] = (__bf16)v;
    }
    for (int i = g; i < B_*KMAC; i += gs) {
      int b = i / KMAC, k = i % KMAC;
      float v = 0.f;
      if (k < A_*MD_) v = mbuf[((size_t)(k / MD_)*B_ + b)*MDP + (k % MD_)];
      actmac[i] = (__bf16)v;
    }
    gsync(bar, nwg);

    // P6: hp1; gi_mac; gh_mac (independent)
    gemm_tiles<true>(actpr, (size_t)B_*KPR, wb + BO_WPR1, (size_t)HDP*KPR, p.bpr1, HD_,
                     hp1, (size_t)B_*HDP, HDP, KPR, HD_, A_, true, wave, nW, lane);
    gemm_tiles<false>(actmac, 0, wb + BO_WIHMAC, 0, p.bih_mac, 0,
                      gimac, 0, G3, KMAC, 3*HMAC_, 1, false, wave, nW, lane);
    gemm_tiles<false>(hmacS, 0, wb + BO_WHHMAC, 0, p.bhh_mac, 0,
                      ghmac, 0, G3, HDP, 3*HMAC_, 1, false, wave, nW, lane);
    gsync(bar, nwg);

    // P7: hp = relu(hp1 @ Wpr2 + bpr2)
    gemm_tiles<true>(hp1, (size_t)B_*HDP, wb + BO_WPR2, (size_t)HDP*HDP, p.bpr2, HD_,
                     hp, (size_t)B_*HDP, HDP, HDP, HD_, A_, true, wave, nW, lane);
    gsync(bar, nwg);

    // P8: fused z GEMM (zc = hp @ [Wpm|Wps] + [bpm|bps]) + macro-GRU combine
    gemm_tiles<false>(hp, (size_t)B_*HDP, wb + BO_WZC, (size_t)32*HDP, biasZ, 32,
                      zc, (size_t)B_*32, 32, HDP, 32, A_, false, wave, nW, lane);
    for (int i = g; i < B_*HDP; i += gs) {
      int b = i / HDP, j = i % HDP;
      float hn = 0.f;
      if (j < HMAC_) {
        const float* gi = gimac + (size_t)b*G3;
        const float* gh = ghmac + (size_t)b*G3;
        float r  = sigmoidf_(gi[j] + gh[j]);
        float zg = sigmoidf_(gi[HMAC_ + j] + gh[HMAC_ + j]);
        float nn = tanhf(gi[2*HMAC_ + j] + r * gh[2*HMAC_ + j]);
        float h  = hmac[(size_t)b*HDP + j];
        hn = (1.f - zg) * nn + zg * h;
      }
      hmacN[i]  = hn;
      hmacNS[i] = (__bf16)hn;
    }
    gsync(bar, nwg);

    // P9: z = zc.mean + softplus(zc.std) * eps_z
    for (int i = g; i < A_*B_*ZDP; i += gs) {
      int ab = i / ZDP, j = i % ZDP;
      float v = 0.f;
      if (j < ZD_) {
        int a = ab / B_, b = ab % B_;
        float ez = p.eps_z[(((size_t)t*A_ + a)*B_ + b)*ZD_ + j];
        v = zc[(size_t)ab*32 + j] + softplusf_(zc[(size_t)ab*32 + 16 + j]) * ez;
      }
      zbuf[i] = v;
    }
    gsync(bar, nwg);

    // P10: actd = [y | m | z | h_mic]  (bf16)
    for (int i = g; i < A_*B_*KD_IN; i += gs) {
      int a = i / (B_*KD_IN); int r = i % (B_*KD_IN);
      int b = r / KD_IN, k = r % KD_IN;
      size_t ab = (size_t)a*B_ + b;
      float v = 0.f;
      if (k < YD_)                       v = yb[b*32 + k];
      else if (k < YD_+MD_)              v = mbuf[ab*MDP + (k - YD_)];
      else if (k < YD_+MD_+ZD_)          v = zbuf[ab*ZDP + (k - YD_ - MD_)];
      else if (k < YD_+MD_+ZD_+HMIC_)    v = hmic[ab*HDP + (k - YD_ - MD_ - ZD_)];
      actd[i] = (__bf16)v;
    }
    gsync(bar, nwg);

    // P11: hd1
    gemm_tiles<true>(actd, (size_t)B_*KD_IN, wb + BO_WD1, (size_t)HDP*KD_IN, p.bd1, HD_,
                     hd1, (size_t)B_*HDP, HDP, KD_IN, HD_, A_, true, wave, nW, lane);
    gsync(bar, nwg);

    // P12: hd
    gemm_tiles<true>(hd1, (size_t)B_*HDP, wb + BO_WD2, (size_t)HDP*HDP, p.bd2, HD_,
                     hdb, (size_t)B_*HDP, HDP, HDP, HD_, A_, true, wave, nW, lane);
    gsync(bar, nwg);

    // P13: fused x GEMM (xc = hd @ [Wdx|Wds] + [bdx|bds])
    gemm_tiles<false>(hdb, (size_t)B_*HDP, wb + BO_WXC, (size_t)32*HDP, biasX, 32,
                      xc, (size_t)B_*32, 32, HDP, 32, A_, false, wave, nW, lane);
    gsync(bar, nwg);

    // P14: x sample / teacher-force, y_next assembly + ret output, actmic
    for (int i = g; i < A_*B_; i += gs) {
      int a = i / B_, b = i % B_;
      size_t ab = (size_t)a*B_ + b;
      float xv[XD_];
#pragma unroll
      for (int d = 0; d < XD_; ++d) {
        float xsamp = xc[ab*32 + d] + softplusf_(xc[ab*32 + 16 + d]) *
                      p.eps_x[(((size_t)t*A_ + a)*B_ + b)*XD_ + d];
        float xtrue = p.y[((size_t)(t+1)*B_ + b)*YD_ + a*XD_ + d];
        float x = (t < burn) ? xtrue : xsamp;
        xv[d] = x;
        ybN[b*32 + a*XD_ + d] = x;
        p.out[((size_t)(t+1)*B_ + b)*YD_ + a*XD_ + d] = x;
      }
      if (a == 0)
        for (int k = YD_; k < 32; ++k) ybN[b*32 + k] = 0.f;
      __bf16* mic = actmic + ab*KMIC;
      mic[0] = (__bf16)xv[0]; mic[1] = (__bf16)xv[1];
      for (int k = 0; k < ZD_; ++k) mic[2 + k] = (__bf16)zbuf[ab*ZDP + k];
      for (int k = 2 + ZD_; k < KMIC; ++k) mic[k] = (__bf16)0.f;
    }
    gsync(bar, nwg);

    // P15: gi_mic, gh_mic
    gemm_tiles<false>(actmic, (size_t)B_*KMIC, wb + BO_WIHMIC, (size_t)G3*KMIC, p.bih_mic, 3*HMIC_,
                      gimic, (size_t)B_*G3, G3, KMIC, 3*HMIC_, A_, false, wave, nW, lane);
    gemm_tiles<false>(hmicS, (size_t)B_*HDP, wb + BO_WHHMIC, (size_t)G3*HDP, p.bhh_mic, 3*HMIC_,
                      ghmic, (size_t)B_*G3, G3, HDP, 3*HMIC_, A_, false, wave, nW, lane);
    gsync(bar, nwg);

    // P16: micro-GRU combine
    for (int i = g; i < A_*B_*HDP; i += gs) {
      int ab = i / HDP, j = i % HDP;
      float hn = 0.f;
      if (j < HMIC_) {
        const float* gi = gimic + (size_t)ab*G3;
        const float* gh = ghmic + (size_t)ab*G3;
        float r  = sigmoidf_(gi[j] + gh[j]);
        float zg = sigmoidf_(gi[HMIC_ + j] + gh[HMIC_ + j]);
        float nn = tanhf(gi[2*HMIC_ + j] + r * gh[2*HMIC_ + j]);
        float h  = hmic[i];
        hn = (1.f - zg) * nn + zg * h;
      }
      hmicN[i]  = hn;
      hmicNS[i] = (__bf16)hn;
    }
    gsync(bar, nwg);
  }
}

// ---------------------------------------------------------------------------
extern "C" void kernel_launch(void* const* d_in, const int* in_sizes, int n_in,
                              void* d_out, int out_size, void* d_ws, size_t ws_size,
                              hipStream_t stream) {
  (void)in_sizes; (void)n_in; (void)out_size; (void)ws_size;
  Params p;
  p.y       = (const float*)d_in[0];
  p.mg      = (const int*)d_in[1];
  p.Wdm1    = (const float*)d_in[2];  p.bdm1 = (const float*)d_in[3];
  p.Wdm2    = (const float*)d_in[4];  p.bdm2 = (const float*)d_in[5];
  p.Wpr1    = (const float*)d_in[6];  p.bpr1 = (const float*)d_in[7];
  p.Wpr2    = (const float*)d_in[8];  p.bpr2 = (const float*)d_in[9];
  p.Wpm     = (const float*)d_in[10]; p.bpm  = (const float*)d_in[11];
  p.Wps     = (const float*)d_in[12]; p.bps  = (const float*)d_in[13];
  p.Wd1     = (const float*)d_in[14]; p.bd1  = (const float*)d_in[15];
  p.Wd2     = (const float*)d_in[16]; p.bd2  = (const float*)d_in[17];
  p.Wdx     = (const float*)d_in[18]; p.bdx  = (const float*)d_in[19];
  p.Wds     = (const float*)d_in[20]; p.bds  = (const float*)d_in[21];
  p.Wih_mic = (const float*)d_in[22]; p.Whh_mic = (const float*)d_in[23];
  p.bih_mic = (const float*)d_in[24]; p.bhh_mic = (const float*)d_in[25];
  p.Wih_mac = (const float*)d_in[26]; p.Whh_mac = (const float*)d_in[27];
  p.bih_mac = (const float*)d_in[28]; p.bhh_mac = (const float*)d_in[29];
  p.eps_g   = (const float*)d_in[30];
  p.eps_z   = (const float*)d_in[31];
  p.eps_x   = (const float*)d_in[32];
  p.burn    = (const int*)d_in[33];
  p.out     = (float*)d_out;

  char* base = (char*)d_ws;
  p.wsf = (float*)base;
  size_t fb = (F_TOTAL * sizeof(float) + 255) & ~(size_t)255;
  p.wsb = (unsigned short*)(base + fb);
  size_t bb = (B_TOTAL * sizeof(unsigned short) + 255) & ~(size_t)255;
  p.bar = (unsigned*)(base + fb + bb);

  vrnn_prep<<<PREP_BLOCKS, BLK, 0, stream>>>(p);
  vrnn_main<<<NWG, BLK, 0, stream>>>(p);
}
